// roadLoss_33234456937175
// MI455X (gfx1250) — matrix operand order repaired and checked by reading, not compiled, not run on hardware
//
#include <hip/hip_runtime.h>
#include <stdint.h>

// ---------------------------------------------------------------------------
// Topology loss on MI455X (gfx1250, wave32).
//   - Bit-packed Zhang-Suen thinning (bit-sliced carry-save adders): the
//     iterative core stays L2-resident (1MB/buffer) and VALU-bound.
//   - 5x5 box-sum "buffer" conv expressed as banded GEMMs per 16x16 tile,
//     executed with v_wmma_f32_16x16x32_f16 (2 WMMAs for X*C, LDS relayout,
//     1 WMMA for R*Y), masked-accumulated straight out of D-layout.
//   - Integer accumulators -> bitwise-deterministic across graph replays.
// ---------------------------------------------------------------------------

#define WORDS_PER_IMG 32768   // 1024 rows * 32 u32 words
#define N_IMGS 8
#define TOT_WORDS (N_IMGS * WORDS_PER_IMG)   // 262144

typedef __attribute__((ext_vector_type(16))) _Float16 v16h;
typedef __attribute__((ext_vector_type(8)))  float    v8f;

// ---------------- bit helpers -----------------------------------------------

__device__ __forceinline__ uint32_t getbit(const uint32_t* __restrict__ img, int r, int c) {
  if ((unsigned)r >= 1024u || (unsigned)c >= 1024u) return 0u;
  return (img[r * 32 + (c >> 5)] >> (c & 31)) & 1u;
}

struct NB { uint32_t P2, P3, P4, P5, P6, P7, P8, P9, cur; };

__device__ __forceinline__ uint32_t ldw(const uint32_t* __restrict__ b, int r, int w) {
  return ((unsigned)r < 1024u && (unsigned)w < 32u) ? b[r * 32 + w] : 0u;
}

// Gather the 8 neighbor bit-planes for one 32-pixel word (bit i = column).
__device__ __forceinline__ NB gatherNB(const uint32_t* __restrict__ buf, int img, int r, int w) {
  const uint32_t* b = buf + img * WORDS_PER_IMG;
  uint32_t upL = ldw(b, r - 1, w - 1), up = ldw(b, r - 1, w), upR = ldw(b, r - 1, w + 1);
  uint32_t cL  = ldw(b, r,     w - 1), cu = ldw(b, r,     w), cR  = ldw(b, r,     w + 1);
  uint32_t dnL = ldw(b, r + 1, w - 1), dn = ldw(b, r + 1, w), dnR = ldw(b, r + 1, w + 1);
  NB n;
  n.cur = cu;
  n.P2 = up;                          // N
  n.P6 = dn;                          // S
  n.P4 = (cu >> 1) | (cR << 31);      // E  (bit i <- col i+1)
  n.P8 = (cu << 1) | (cL >> 31);      // W
  n.P3 = (up >> 1) | (upR << 31);     // NE
  n.P9 = (up << 1) | (upL >> 31);     // NW
  n.P5 = (dn >> 1) | (dnR << 31);     // SE
  n.P7 = (dn << 1) | (dnL >> 31);     // SW
  return n;
}

__device__ __forceinline__ void fa(uint32_t a, uint32_t b, uint32_t c,
                                   uint32_t& s, uint32_t& cy) {
  uint32_t t = a ^ b; s = t ^ c; cy = (a & b) | (t & c);
}

// Bit-sliced popcount of 8 one-bit planes -> 4 bit-planes b3b2b1b0 (0..8).
__device__ __forceinline__ void count8(uint32_t x0, uint32_t x1, uint32_t x2, uint32_t x3,
                                       uint32_t x4, uint32_t x5, uint32_t x6, uint32_t x7,
                                       uint32_t& b0, uint32_t& b1, uint32_t& b2, uint32_t& b3) {
  uint32_t s1, c1, s2, c2, s3, c3;
  fa(x0, x1, x2, s1, c1);
  fa(x3, x4, x5, s2, c2);
  fa(s1, s2, x6, s3, c3);
  uint32_t c4 = s3 & x7; b0 = s3 ^ x7;
  uint32_t t1, d1; fa(c1, c2, c3, t1, d1);
  uint32_t d2 = t1 & c4; b1 = t1 ^ c4;
  b2 = d1 ^ d2; b3 = d1 & d2;
}

// ---------------- kernels ---------------------------------------------------

// pred: round(sigmoid(x)) == 1  <=>  x > 0 ; gt: nonzero int.
__global__ void binarizeKernel(const float* __restrict__ pred, const int* __restrict__ gt,
                               uint32_t* __restrict__ predBits, uint32_t* __restrict__ gtBits) {
  int idx = blockIdx.x * blockDim.x + threadIdx.x;   // one u32 word = 32 pixels
  if (idx >= TOT_WORDS) return;
  long base = (long)idx * 32;
  uint32_t pw = 0u, gw = 0u;
#pragma unroll
  for (int i = 0; i < 32; ++i) {
    pw |= (pred[base + i] > 0.0f ? 1u : 0u) << i;
    gw |= (gt[base + i]   != 0   ? 1u : 0u) << i;
  }
  predBits[idx] = pw;
  gtBits[idx]  = gw;
}

__global__ void initFlags(uint32_t* flags) {
  if (threadIdx.x < 8) flags[threadIdx.x] = 0u;
}

// flags: [0]=active [1]=changed [2]=numerI [3]=numerE [4]=denomI [5]=denomE
__global__ void flagUpdate(uint32_t* flags, int first) {
  flags[0] = first ? 1u : flags[1];
  flags[1] = 0u;
}

// One Zhang-Suen sub-iteration over pred AND gt buffers (bit-sliced, 32 px/lane-word).
__global__ void zsKernel(const uint32_t* __restrict__ inP, uint32_t* __restrict__ outP,
                         const uint32_t* __restrict__ inG, uint32_t* __restrict__ outG,
                         int step, uint32_t* flags) {
  if (flags[0] == 0u) return;          // converged: uniform early-out
  __shared__ uint32_t blk;
  if (threadIdx.x == 0) blk = 0u;
  __syncthreads();

  int gid = blockIdx.x * blockDim.x + threadIdx.x;   // 0 .. 2*TOT_WORDS-1
  const uint32_t* in; uint32_t* out; int idx;
  if (gid < TOT_WORDS) { in = inP; out = outP; idx = gid; }
  else                 { in = inG; out = outG; idx = gid - TOT_WORDS; }
  int img = idx >> 15, widx = idx & 32767, r = widx >> 5, w = widx & 31;

  NB n = gatherNB(in, img, r, w);

  uint32_t b0, b1, b2, b3;
  count8(n.P2, n.P3, n.P4, n.P5, n.P6, n.P7, n.P8, n.P9, b0, b1, b2, b3);
  uint32_t ge2 = b1 | b2 | b3;
  uint32_t le6 = ~(b3 | (b0 & b1 & b2));
  uint32_t inB = ge2 & le6;

  uint32_t t0 = ~n.P2 & n.P3, t1 = ~n.P3 & n.P4, t2 = ~n.P4 & n.P5, t3 = ~n.P5 & n.P6;
  uint32_t t4 = ~n.P6 & n.P7, t5 = ~n.P7 & n.P8, t6 = ~n.P8 & n.P9, t7 = ~n.P9 & n.P2;
  uint32_t a0, a1, a2, a3;
  count8(t0, t1, t2, t3, t4, t5, t6, t7, a0, a1, a2, a3);
  uint32_t aEq1 = a0 & ~a1 & ~a2 & ~a3;

  uint32_t c3m = (step == 0)
      ? (~(n.P2 & n.P4 & n.P6) & ~(n.P4 & n.P6 & n.P8))
      : (~(n.P2 & n.P4 & n.P8) & ~(n.P2 & n.P6 & n.P8));

  uint32_t del = n.cur & inB & aEq1 & c3m;
  out[idx] = n.cur & ~del;
  if (del) atomicOr(&blk, 1u);
  __syncthreads();
  if (threadIdx.x == 0 && blk) atomicOr(&flags[1], 1u);
}

// Degree masks: endpoint = skel & (deg==1); intersection = skel & (deg>=3).
// Optionally accumulate denominators (pred only) into flags[denomBase/+1].
__global__ void maskKernel(const uint32_t* __restrict__ skel,
                           uint32_t* __restrict__ inter, uint32_t* __restrict__ endpt,
                           uint32_t* flags, int denomBase) {
  int idx = blockIdx.x * blockDim.x + threadIdx.x;
  if (idx >= TOT_WORDS) return;
  int img = idx >> 15, widx = idx & 32767, r = widx >> 5, w = widx & 31;
  NB n = gatherNB(skel, img, r, w);
  uint32_t b0, b1, b2, b3;
  count8(n.P2, n.P3, n.P4, n.P5, n.P6, n.P7, n.P8, n.P9, b0, b1, b2, b3);
  uint32_t em = n.cur & (b0 & ~b1 & ~b2 & ~b3);          // deg == 1
  uint32_t im = n.cur & (b3 | b2 | (b1 & b0));           // deg >= 3
  endpt[idx] = em;
  inter[idx] = im;
  if (denomBase >= 0) {
    uint32_t ci = __popc(im), ce = __popc(em);
#pragma unroll
    for (int o = 16; o > 0; o >>= 1) {
      ci += __shfl_xor(ci, o, 32);
      ce += __shfl_xor(ce, o, 32);
    }
    if ((threadIdx.x & 31) == 0) {
      atomicAdd((unsigned int*)&flags[denomBase],     ci);
      atomicAdd((unsigned int*)&flags[denomBase + 1], ce);
    }
  }
}

// 5x5 box-sum of gt mask on a 16x16 tile via banded GEMMs on WMMA:
//   Y = X*C  (20x20 tile * 20x16 band, K padded to 32, rows split 16+4)
//   OUT = R*Y (16x20 band * Y)  -- Y relayed through LDS into B-layout.
// Then masked-sum OUT against the bit-packed pred mask (D-layout direct read).
__global__ void __launch_bounds__(256) convAccumKernel(
    const uint32_t* __restrict__ interG, const uint32_t* __restrict__ endptG,
    const uint32_t* __restrict__ interP, const uint32_t* __restrict__ endptP,
    uint32_t* flags) {
  __shared__ _Float16 ldsY[8][20 * 16];   // per-wave 20x16 Y tile
  int lane  = threadIdx.x & 31;
  int wslot = threadIdx.x >> 5;
  int wave  = blockIdx.x * 8 + wslot;     // 32768 tiles total
  int img = wave >> 12, t = wave & 4095;
  int r0 = (t >> 6) * 16, c0 = (t & 63) * 16;
  int M = lane & 15;                       // A-row / D-col index base
  int N = M;                               // B/D column
  int h = lane >> 4;

  for (int m = 0; m < 2; ++m) {
    const uint32_t* gm = (m == 0 ? interG : endptG) + img * WORDS_PER_IMG;
    const uint32_t* pm = (m == 0 ? interP : endptP) + img * WORDS_PER_IMG;
    __syncthreads();   // LDS tile reusable

    v16h a0 = {}, a1 = {}, bC = {}, aR = {};
#pragma unroll
    for (int v = 0; v < 8; ++v) {
      int Ka = 2 * (v & 3) + 8 * h + 16 * (v >> 2);   // A-layout K (16-bit A 16x32)
      int Kb = 2 * (v & 3) + 8 * h + 16 * (v >> 2);   // B-layout K (16-bit B 32x16)
#pragma unroll
      for (int e = 0; e < 2; ++e) {
        int K = Ka + e;
        a0[2 * v + e] = (K < 20) ? (_Float16)(float)getbit(gm, r0 - 2 + M,  c0 - 2 + K) : (_Float16)0.0f;
        a1[2 * v + e] = (K < 20 && M < 4) ? (_Float16)(float)getbit(gm, r0 + 14 + M, c0 - 2 + K) : (_Float16)0.0f;
        aR[2 * v + e] = (_Float16)((K < 20 && K >= M && K <= M + 4) ? 1.0f : 0.0f);   // R band
        int K2 = Kb + e;
        bC[2 * v + e] = (_Float16)((K2 < 20 && K2 >= N && K2 <= N + 4) ? 1.0f : 0.0f); // C band
      }
    }

    v8f z = {};
    v8f y0 = __builtin_amdgcn_wmma_f32_16x16x32_f16(false, a0, false, bC, (short)0, z, false, false);
    v8f y1 = __builtin_amdgcn_wmma_f32_16x16x32_f16(false, a1, false, bC, (short)0, z, false, false);

    // D-layout -> LDS (row-major 20x16 f16)
#pragma unroll
    for (int v = 0; v < 8; ++v) ldsY[wslot][(v + 8 * h) * 16 + N] = (_Float16)y0[v];
    if (h == 0) {
#pragma unroll
      for (int v = 0; v < 4; ++v) ldsY[wslot][(16 + v) * 16 + N] = (_Float16)y1[v];
    }
    __syncthreads();

    // LDS -> B-layout fragment of Y
    v16h bY = {};
#pragma unroll
    for (int v = 0; v < 8; ++v)
#pragma unroll
      for (int e = 0; e < 2; ++e) {
        int K = 2 * (v & 3) + 8 * h + 16 * (v >> 2) + e;
        bY[2 * v + e] = (K < 20) ? ldsY[wslot][K * 16 + N] : (_Float16)0.0f;
      }

    v8f outT = __builtin_amdgcn_wmma_f32_16x16x32_f16(false, aR, false, bY, (short)0, z, false, false);

    // masked accumulate straight from D-layout (VGPR v, lane l: row v+8h, col N)
    float acc = 0.0f;
#pragma unroll
    for (int v = 0; v < 8; ++v) {
      if (getbit(pm, r0 + v + 8 * h, c0 + N)) acc += outT[v];
    }
#pragma unroll
    for (int o = 16; o > 0; o >>= 1) acc += __shfl_xor(acc, o, 32);
    if (lane == 0) atomicAdd((unsigned int*)&flags[2 + m], (uint32_t)(acc + 0.5f));
  }
}

__global__ void finalizeKernel(const uint32_t* __restrict__ flags, float* __restrict__ out) {
  float Inum = (float)flags[2] / (float)flags[4];
  float Enum = (float)flags[3] / (float)flags[5];
  out[0] = 1.0f - 0.5f * (Inum + Enum);
}

// ---------------- launch ----------------------------------------------------

extern "C" void kernel_launch(void* const* d_in, const int* in_sizes, int n_in,
                              void* d_out, int out_size, void* d_ws, size_t ws_size,
                              hipStream_t stream) {
  (void)in_sizes; (void)n_in; (void)out_size; (void)ws_size;
  const float* pred = (const float*)d_in[0];
  const int*   gt   = (const int*)d_in[1];

  uint32_t* ws     = (uint32_t*)d_ws;          // ~8MB + 32B used
  uint32_t* predA  = ws;
  uint32_t* predB  = ws + 1 * TOT_WORDS;
  uint32_t* gtA    = ws + 2 * TOT_WORDS;
  uint32_t* gtB    = ws + 3 * TOT_WORDS;
  uint32_t* interP = ws + 4 * TOT_WORDS;
  uint32_t* endptP = ws + 5 * TOT_WORDS;
  uint32_t* interG = ws + 6 * TOT_WORDS;
  uint32_t* endptG = ws + 7 * TOT_WORDS;
  uint32_t* flags  = ws + 8 * TOT_WORDS;

  binarizeKernel<<<TOT_WORDS / 256, 256, 0, stream>>>(pred, gt, predA, gtA);
  initFlags<<<1, 8, 0, stream>>>(flags);

  // Upper bound on thinning pairs; converged pairs early-out via flags[0].
  const int PAIRS = 96;
  for (int it = 0; it < PAIRS; ++it) {
    flagUpdate<<<1, 1, 0, stream>>>(flags, it == 0 ? 1 : 0);
    zsKernel<<<2 * TOT_WORDS / 256, 256, 0, stream>>>(predA, predB, gtA, gtB, 0, flags);
    zsKernel<<<2 * TOT_WORDS / 256, 256, 0, stream>>>(predB, predA, gtB, gtA, 1, flags);
  }
  // skeletons now in predA / gtA

  maskKernel<<<TOT_WORDS / 256, 256, 0, stream>>>(predA, interP, endptP, flags, 4);
  maskKernel<<<TOT_WORDS / 256, 256, 0, stream>>>(gtA,   interG, endptG, flags, -1);

  convAccumKernel<<<4096, 256, 0, stream>>>(interG, endptG, interP, endptP, flags);

  finalizeKernel<<<1, 1, 0, stream>>>(flags, (float*)d_out);
}
